// SMILESDecoder_79989470920968
// MI455X (gfx1250) — matrix operand order repaired and checked by reading, not compiled, not run on hardware
//
#include <hip/hip_runtime.h>
#include <hip/hip_bf16.h>

#define BB 8
#define TT 512
#define MM 64
#define DD 1024
#define HH 16
#define DFF 4096
#define DHH 64
#define LL (MM + TT)          // 576
#define NTOK (BB * LL)        // 4608
#define N2 (BB * TT)          // 4096
#define ROT 32

typedef __attribute__((ext_vector_type(16))) _Float16 v16h;
typedef __attribute__((ext_vector_type(8)))  float    v8f;
typedef __attribute__((ext_vector_type(4)))  unsigned u32x4;
typedef __attribute__((ext_vector_type(8)))  int      i32x8;
typedef __attribute__((ext_vector_type(4)))  int      i32x4;

union FragH { v16h h; unsigned u[8]; };

// ------------------------------------------------------------------ TDM issue
// 2D tile load Global->LDS via Tensor Data Mover (D# per ISA 08 §8.3/8.4).
// LDS rows get padded: 16 data DWORDs + 4 pad DWORDs -> 40 f16 (80B) row pitch,
// matching the conflict-free fragment-gather layout (20-bank stride).
__device__ __forceinline__ void tdm_load_2d_f16(unsigned ldsOff, const void* gptr,
                                                unsigned tileD0, unsigned tileD1,
                                                unsigned tensorD0, unsigned tensorD1,
                                                unsigned long long stride0) {
  unsigned long long ga = (unsigned long long)(uintptr_t)gptr;
  u32x4 g0;
  g0[0] = 1u;                                            // count=1, user D#
  g0[1] = ldsOff;                                        // lds byte address
  g0[2] = (unsigned)ga;                                  // global_addr[31:0]
  g0[3] = (unsigned)((ga >> 32) & 0x1FFFFFFull) | (2u << 30); // [56:32] | type=2
  i32x8 g1;
  // data_size=1 (2B) | pad_enable | pad_interval=3 (16 dw) | pad_amount=3 (4 dw)
  g1[0] = (int)((1u << 16) | (1u << 20) | (3u << 22) | (3u << 25));
  g1[1] = (int)((tensorD0 & 0xFFFFu) << 16);             // atomic_barrier=0
  g1[2] = (int)(((tensorD0 >> 16) & 0xFFFFu) | ((tensorD1 & 0xFFFFu) << 16));
  g1[3] = (int)(((tensorD1 >> 16) & 0xFFFFu) | (tileD0 << 16));
  g1[4] = (int)(tileD1 & 0xFFFFu);                       // tile_dim1 | tile_dim2=0
  g1[5] = (int)(unsigned)(stride0 & 0xFFFFFFFFull);      // tensor_dim0_stride lo
  g1[6] = (int)(unsigned)((stride0 >> 32) & 0xFFFFull);  // stride hi | d1stride=0
  g1[7] = 0;
  i32x4 z4 = {0, 0, 0, 0};
  i32x8 z8 = {0, 0, 0, 0, 0, 0, 0, 0};
  // 6-arg form (clang-23 / therock headers): (g0, g1, g2, g3, g4, cpol)
  __builtin_amdgcn_tensor_load_to_lds(g0, g1, z4, z4, z8, 0);
}

// ---------------------------------------------------------------- utilities
__global__ void f32_to_f16_kernel(const float* __restrict__ in,
                                  _Float16* __restrict__ out, size_t n) {
  size_t i = (size_t)blockIdx.x * blockDim.x + threadIdx.x;
  if (i < n) out[i] = (_Float16)in[i];
}

// ------------------------------------------------- concat(memory,x) + LN -> f16
__global__ __launch_bounds__(256)
void concat_ln_kernel(const float* __restrict__ x, const float* __restrict__ mem,
                      const float* __restrict__ g, const float* __restrict__ be,
                      float* __restrict__ xfull, _Float16* __restrict__ xa) {
  __shared__ float red[8];
  int row = blockIdx.x, tid = threadIdx.x;
  int b = row / LL, l = row % LL;
  const float* src = (l < MM) ? (mem + ((size_t)b * MM + l) * DD)
                              : (x + ((size_t)b * TT + (l - MM)) * DD);
  float v[4]; float s = 0.f;
#pragma unroll
  for (int i = 0; i < 4; ++i) { v[i] = src[tid + i * 256]; s += v[i]; }
#pragma unroll
  for (int off = 16; off > 0; off >>= 1) s += __shfl_xor(s, off, 32);
  if ((tid & 31) == 0) red[tid >> 5] = s;
  __syncthreads();
  float tot = 0.f;
#pragma unroll
  for (int i = 0; i < 8; ++i) tot += red[i];
  float mu = tot * (1.f / (float)DD);
  __syncthreads();
  float vs = 0.f;
#pragma unroll
  for (int i = 0; i < 4; ++i) { float d = v[i] - mu; vs += d * d; }
#pragma unroll
  for (int off = 16; off > 0; off >>= 1) vs += __shfl_xor(vs, off, 32);
  if ((tid & 31) == 0) red[tid >> 5] = vs;
  __syncthreads();
  float vtot = 0.f;
#pragma unroll
  for (int i = 0; i < 8; ++i) vtot += red[i];
  float rstd = rsqrtf(vtot * (1.f / (float)DD) + 1e-5f);
  float* df = xfull + (size_t)row * DD;
  _Float16* dh = xa + (size_t)row * DD;
#pragma unroll
  for (int i = 0; i < 4; ++i) {
    int c = tid + i * 256;
    df[c] = v[i];
    dh[c] = (_Float16)((v[i] - mu) * rstd * g[c] + be[c]);
  }
}

// --------------------------------------------- residual add + LN (post-attn)
__global__ __launch_bounds__(256)
void residual_ln_kernel(const float* __restrict__ y, const float* __restrict__ xfull,
                        const float* __restrict__ g, const float* __restrict__ be,
                        float* __restrict__ x2, _Float16* __restrict__ xn) {
  __shared__ float red[8];
  int row = blockIdx.x, tid = threadIdx.x;     // row in [0, N2)
  int b = row / TT, t = row % TT;
  const float* xr = xfull + ((size_t)(b * LL + MM + t)) * DD;
  const float* yr = y + (size_t)row * DD;
  float v[4]; float s = 0.f;
#pragma unroll
  for (int i = 0; i < 4; ++i) {
    int c = tid + i * 256;
    v[i] = xr[c] + yr[c]; s += v[i];
  }
#pragma unroll
  for (int off = 16; off > 0; off >>= 1) s += __shfl_xor(s, off, 32);
  if ((tid & 31) == 0) red[tid >> 5] = s;
  __syncthreads();
  float tot = 0.f;
#pragma unroll
  for (int i = 0; i < 8; ++i) tot += red[i];
  float mu = tot * (1.f / (float)DD);
  __syncthreads();
  float vs = 0.f;
#pragma unroll
  for (int i = 0; i < 4; ++i) { float d = v[i] - mu; vs += d * d; }
#pragma unroll
  for (int off = 16; off > 0; off >>= 1) vs += __shfl_xor(vs, off, 32);
  if ((tid & 31) == 0) red[tid >> 5] = vs;
  __syncthreads();
  float vtot = 0.f;
#pragma unroll
  for (int i = 0; i < 8; ++i) vtot += red[i];
  float rstd = rsqrtf(vtot * (1.f / (float)DD) + 1e-5f);
  float* x2r = x2 + (size_t)row * DD;
  _Float16* xnr = xn + (size_t)row * DD;
#pragma unroll
  for (int i = 0; i < 4; ++i) {
    int c = tid + i * 256;
    x2r[c] = v[i];
    xnr[c] = (_Float16)((v[i] - mu) * rstd * g[c] + be[c]);
  }
}

// ----------------------------- generic WMMA GEMM: out = A[M,K] * W[N,K]^T + b
// TDM double-buffered LDS staging; 128x64 block tile, 8 waves, K-step 32.
// OUT: 0 -> f32 out (+ optional same-shape f32 residual), 1 -> f16 out
// ACT: 0 none, 1 squared-relu
template <int OUT, int ACT>
__global__ __launch_bounds__(256)
void gemm_wmma_kernel(const _Float16* __restrict__ A, const _Float16* __restrict__ W,
                      const float* __restrict__ bias, const float* __restrict__ res,
                      void* __restrict__ out, int Mrows, int N, int K) {
  __shared__ _Float16 Alds[2][128 * 40];
  __shared__ _Float16 Blds[2][64 * 40];
  const int tid = threadIdx.x;
  const int lane = tid & 31, wid = tid >> 5;
  const int lmod = lane & 15, lhi = lane >> 4;
  const int nBlkN = N >> 6;
  const int m0 = (blockIdx.x / nBlkN) * 128;
  const int n0 = (blockIdx.x % nBlkN) * 64;
  const int nsteps = K >> 5;
  const bool issuer = (wid == 0);
  v8f acc[4] = {};

  if (issuer) {
    tdm_load_2d_f16((unsigned)(uintptr_t)&Alds[0][0], A + (size_t)m0 * K,
                    32u, 128u, (unsigned)K, (unsigned)(Mrows - m0), (unsigned long long)K);
    tdm_load_2d_f16((unsigned)(uintptr_t)&Blds[0][0], W + (size_t)n0 * K,
                    32u, 64u, (unsigned)K, (unsigned)(N - n0), (unsigned long long)K);
  }
  for (int i = 0; i < nsteps; ++i) {
    const int cur = i & 1;
    if (issuer) {
      if (i + 1 < nsteps) {
        const int k1 = (i + 1) << 5;
        tdm_load_2d_f16((unsigned)(uintptr_t)&Alds[cur ^ 1][0],
                        A + (size_t)m0 * K + k1, 32u, 128u,
                        (unsigned)(K - k1), (unsigned)(Mrows - m0), (unsigned long long)K);
        tdm_load_2d_f16((unsigned)(uintptr_t)&Blds[cur ^ 1][0],
                        W + (size_t)n0 * K + k1, 32u, 64u,
                        (unsigned)(K - k1), (unsigned)(N - n0), (unsigned long long)K);
        __builtin_amdgcn_s_wait_tensorcnt(2);  // retire current tile's pair
      } else {
        __builtin_amdgcn_s_wait_tensorcnt(0);
      }
    }
    __syncthreads();
    FragH af;
    const _Float16* Ab = &Alds[cur][0];
    const _Float16* Bb = &Blds[cur][0];
    const int am = (wid * 16 + lmod) * 40;
#pragma unroll
    for (int j = 0; j < 8; ++j) {
      int kl = (j >> 2) * 16 + lhi * 8 + (j & 3) * 2;
      af.u[j] = *(const unsigned*)(Ab + am + kl);
    }
#pragma unroll
    for (int t = 0; t < 4; ++t) {
      FragH bf;
      const int bnr = (t * 16 + lmod) * 40 + lhi * 8;
#pragma unroll
      for (int j = 0; j < 8; ++j)
        bf.u[j] = *(const unsigned*)(Bb + bnr + (j >> 2) * 16 + (j & 3) * 2);
      acc[t] = __builtin_amdgcn_wmma_f32_16x16x32_f16(false, af.h, false, bf.h,
                                                      (short)0, acc[t], false, false);
    }
    __syncthreads();   // all waves done with buf[cur] before TDM overwrites it
  }
#pragma unroll
  for (int t = 0; t < 4; ++t) {
    int col = n0 + t * 16 + lmod;
    float bv = bias[col];
#pragma unroll
    for (int r = 0; r < 8; ++r) {
      int row = m0 + wid * 16 + r + 8 * lhi;
      float v = acc[t][r] + bv;
      if (ACT == 1) { v = v > 0.f ? v * v : 0.f; }
      if (res) v += res[(size_t)row * N + col];
      if (OUT == 0) ((float*)out)[(size_t)row * N + col] = v;
      else ((_Float16*)out)[(size_t)row * N + col] = (_Float16)v;
    }
  }
}

// ---------------------------- token-major [NTOK,D] f16 -> head-major (+RoPE)
// mode 0: q (RoPE, *1/sqrt(DH)) -> [B,H,L,DH]; mode 1: k (RoPE) -> [B,H,L,DH]
// mode 2: v -> transposed [B,H,DH,L]
__global__ void reshape_qkv_kernel(const _Float16* __restrict__ in,
                                   _Float16* __restrict__ out, int mode) {
  size_t i = (size_t)blockIdx.x * blockDim.x + threadIdx.x;
  if (i >= (size_t)NTOK * DD) return;
  int c = (int)(i % DD); int row = (int)(i / DD);
  int b = row / LL, l = row % LL;
  int h = c / DHH, dh = c % DHH;
  float v = (float)in[i];
  if (mode <= 1) {
    if (dh < ROT) {
      int f = dh & 15;
      float invf = __powf(10000.f, -(float)f / 16.f);
      float ang = (float)l * invf;
      float cs = __cosf(ang), sn = __sinf(ang);
      float other = (float)in[(size_t)row * DD + h * DHH + (dh < 16 ? dh + 16 : dh - 16)];
      v = (dh < 16) ? (v * cs - other * sn) : (v * cs + other * sn);
    }
    if (mode == 0) v *= 0.125f;   // 1/sqrt(64) folded into q
    out[(((size_t)b * HH + h) * LL + l) * DHH + dh] = (_Float16)v;
  } else {
    out[(((size_t)b * HH + h) * DHH + dh) * LL + l] = (_Float16)v;
  }
}

// -------------------------------------------------- flash attention via WMMA
// q,k: [B,H,L,DH] f16 (q pre-scaled); vT: [B,H,DH,L] f16; o: [B,T,D] f16
__global__ __launch_bounds__(128)
void attention_kernel(const _Float16* __restrict__ qg, const _Float16* __restrict__ kg,
                      const _Float16* __restrict__ vtg, _Float16* __restrict__ og) {
  __shared__ _Float16 Plds[4][16 * 32];
  const int wid = threadIdx.x >> 5, lane = threadIdx.x & 31;
  const int lmod = lane & 15, lhi = lane >> 4;
  const int gtile = blockIdx.x * 4 + wid;          // 0 .. B*H*(T/16)-1
  const int bh = gtile / (TT / 16);
  const int tq = gtile % (TT / 16);
  const _Float16* qbase = qg + (size_t)bh * LL * DHH;
  const _Float16* kbase = kg + (size_t)bh * LL * DHH;
  const _Float16* vbase = vtg + (size_t)bh * DHH * LL;
  const int qr0 = MM + tq * 16;

  FragH qa[2];
#pragma unroll
  for (int h2 = 0; h2 < 2; ++h2) {
    const _Float16* qrow = qbase + (size_t)(qr0 + lmod) * DHH + h2 * 32;
#pragma unroll
    for (int j = 0; j < 8; ++j) {
      int kl = (j >> 2) * 16 + lhi * 8 + (j & 3) * 2;
      qa[h2].u[j] = *(const unsigned*)(qrow + kl);
    }
  }
  v8f oacc[4] = {};
  float mrow[8], lrow[8];
#pragma unroll
  for (int r = 0; r < 8; ++r) { mrow[r] = -1e30f; lrow[r] = 0.f; }
  const int kend = ((qr0 + 15) / 32 + 1) * 32;
  _Float16* P = Plds[wid];

  for (int c0 = 0; c0 < kend; c0 += 32) {
    v8f st[2];
#pragma unroll
    for (int kt = 0; kt < 2; ++kt) {
      v8f s = {};
#pragma unroll
      for (int h2 = 0; h2 < 2; ++h2) {
        FragH bf;
        const _Float16* krow = kbase + (size_t)(c0 + kt * 16 + lmod) * DHH + h2 * 32 + lhi * 8;
#pragma unroll
        for (int j = 0; j < 8; ++j) bf.u[j] = *(const unsigned*)(krow + (j >> 2) * 16 + (j & 3) * 2);
        s = __builtin_amdgcn_wmma_f32_16x16x32_f16(false, qa[h2].h, false, bf.h,
                                                   (short)0, s, false, false);
      }
      st[kt] = s;
    }
    // Only the final 32-wide tile can straddle the causal diagonal
    // (qr0 >= M and qr0 mod 32 in {0,16} => earlier tiles fully allowed).
    const bool maskTile = (c0 + 32 == kend);
#pragma unroll
    for (int r = 0; r < 8; ++r) {
      if (maskTile) {
        int qr = qr0 + r + 8 * lhi;
#pragma unroll
        for (int kt = 0; kt < 2; ++kt) {
          int col = c0 + kt * 16 + lmod;
          bool allowed = (col < MM) || (col <= qr);
          if (!allowed) st[kt][r] = -1e30f;
        }
      }
      float mx = fmaxf(st[0][r], st[1][r]);
#pragma unroll
      for (int off = 1; off < 16; off <<= 1) mx = fmaxf(mx, __shfl_xor(mx, off, 32));
      float mnew = fmaxf(mrow[r], mx);
      float scale = __expf(mrow[r] - mnew);
      mrow[r] = mnew;
      float p0 = __expf(st[0][r] - mnew);
      float p1 = __expf(st[1][r] - mnew);
      float ps = p0 + p1;
#pragma unroll
      for (int off = 1; off < 16; off <<= 1) ps += __shfl_xor(ps, off, 32);
      lrow[r] = lrow[r] * scale + ps;
#pragma unroll
      for (int t = 0; t < 4; ++t) oacc[t][r] *= scale;
      int prow = (r + 8 * lhi) * 32;
      P[prow + lmod] = (_Float16)p0;
      P[prow + 16 + lmod] = (_Float16)p1;
    }
    asm volatile("s_wait_dscnt 0" ::: "memory");
    FragH pa;
    {
      const _Float16* prow = P + lmod * 32;
#pragma unroll
      for (int j = 0; j < 8; ++j) {
        int kl = (j >> 2) * 16 + lhi * 8 + (j & 3) * 2;
        pa.u[j] = *(const unsigned*)(prow + kl);
      }
    }
#pragma unroll
    for (int t = 0; t < 4; ++t) {
      FragH bf;
      const _Float16* vrow = vbase + (size_t)(t * 16 + lmod) * LL + c0 + lhi * 8;
#pragma unroll
      for (int j = 0; j < 8; ++j) bf.u[j] = *(const unsigned*)(vrow + (j >> 2) * 16 + (j & 3) * 2);
      oacc[t] = __builtin_amdgcn_wmma_f32_16x16x32_f16(false, pa.h, false, bf.h,
                                                       (short)0, oacc[t], false, false);
    }
  }
  const int b = bh / HH, h = bh % HH;
#pragma unroll
  for (int r = 0; r < 8; ++r) {
    int trow = tq * 16 + r + 8 * lhi;
    float inv = 1.f / lrow[r];
    _Float16* orow = og + ((size_t)(b * TT + trow)) * DD + h * DHH;
#pragma unroll
    for (int t = 0; t < 4; ++t)
      orow[t * 16 + lmod] = (_Float16)(oacc[t][r] * inv);
  }
}

// =========================================================== host launcher
extern "C" void kernel_launch(void* const* d_in, const int* in_sizes, int n_in,
                              void* d_out, int out_size, void* d_ws, size_t ws_size,
                              hipStream_t stream) {
  (void)in_sizes; (void)n_in; (void)out_size; (void)ws_size;
  const float* x      = (const float*)d_in[0];
  const float* mem    = (const float*)d_in[1];
  const float* Wq     = (const float*)d_in[2];
  const float* bq     = (const float*)d_in[3];
  const float* Wk     = (const float*)d_in[4];
  const float* bk     = (const float*)d_in[5];
  const float* Wv     = (const float*)d_in[6];
  const float* bv     = (const float*)d_in[7];
  const float* Wo     = (const float*)d_in[8];
  const float* bo     = (const float*)d_in[9];
  const float* W1     = (const float*)d_in[10];
  const float* b1     = (const float*)d_in[11];
  const float* W2     = (const float*)d_in[12];
  const float* b2     = (const float*)d_in[13];
  const float* g_attn = (const float*)d_in[14];
  const float* be_attn= (const float*)d_in[15];
  const float* g_mlp  = (const float*)d_in[16];
  const float* be_mlp = (const float*)d_in[17];

  char* ws = (char*)d_ws;
  size_t off = 0;
  auto alloc = [&](size_t bytes) { size_t o = off; off += (bytes + 255) & ~(size_t)255; return o; };
  const size_t o_xfull = alloc((size_t)NTOK * DD * 4);
  const size_t o_xa    = alloc((size_t)NTOK * DD * 2);
  const size_t o_wq    = alloc((size_t)DD * DD * 2);
  const size_t o_wk    = alloc((size_t)DD * DD * 2);
  const size_t o_wv    = alloc((size_t)DD * DD * 2);
  const size_t o_wo    = alloc((size_t)DD * DD * 2);
  const size_t o_w1    = alloc((size_t)DFF * DD * 2);
  const size_t o_w2    = alloc((size_t)DD * DFF * 2);
  const size_t o_qtmp  = alloc((size_t)NTOK * DD * 2);
  const size_t o_qbh   = alloc((size_t)NTOK * DD * 2);
  const size_t o_kbh   = alloc((size_t)NTOK * DD * 2);
  const size_t o_vt    = alloc((size_t)NTOK * DD * 2);
  const size_t o_oh    = alloc((size_t)N2 * DD * 2);
  const size_t o_h1    = alloc((size_t)N2 * DFF * 2);
  // aliases over dead regions
  const size_t o_yattn = o_qbh;   // f32 N2*D  (<= qbh+kbh)
  const size_t o_x2    = o_vt;    // f32 N2*D  (<= vt+oh)
  const size_t o_xn    = o_xa;    // f16 N2*D  (<= xa)

  float*    xfull = (float*)(ws + o_xfull);
  _Float16* xa    = (_Float16*)(ws + o_xa);
  _Float16* wqh   = (_Float16*)(ws + o_wq);
  _Float16* wkh   = (_Float16*)(ws + o_wk);
  _Float16* wvh   = (_Float16*)(ws + o_wv);
  _Float16* woh   = (_Float16*)(ws + o_wo);
  _Float16* w1h   = (_Float16*)(ws + o_w1);
  _Float16* w2h   = (_Float16*)(ws + o_w2);
  _Float16* qtmp  = (_Float16*)(ws + o_qtmp);
  _Float16* qbh   = (_Float16*)(ws + o_qbh);
  _Float16* kbh   = (_Float16*)(ws + o_kbh);
  _Float16* vt    = (_Float16*)(ws + o_vt);
  _Float16* oh    = (_Float16*)(ws + o_oh);
  _Float16* h1    = (_Float16*)(ws + o_h1);
  float*    yattn = (float*)(ws + o_yattn);
  float*    x2    = (float*)(ws + o_x2);
  _Float16* xn    = (_Float16*)(ws + o_xn);

  // 1) weights -> f16 (L2-resident workspace)
  const size_t nW = (size_t)DD * DD, nW1 = (size_t)DFF * DD;
  f32_to_f16_kernel<<<(nW + 255) / 256, 256, 0, stream>>>(Wq, wqh, nW);
  f32_to_f16_kernel<<<(nW + 255) / 256, 256, 0, stream>>>(Wk, wkh, nW);
  f32_to_f16_kernel<<<(nW + 255) / 256, 256, 0, stream>>>(Wv, wvh, nW);
  f32_to_f16_kernel<<<(nW + 255) / 256, 256, 0, stream>>>(Wo, woh, nW);
  f32_to_f16_kernel<<<(nW1 + 255) / 256, 256, 0, stream>>>(W1, w1h, nW1);
  f32_to_f16_kernel<<<(nW1 + 255) / 256, 256, 0, stream>>>(W2, w2h, nW1);

  // 2) concat + LN
  concat_ln_kernel<<<NTOK, 256, 0, stream>>>(x, mem, g_attn, be_attn, xfull, xa);

  const int rshGrid = (int)(((size_t)NTOK * DD + 255) / 256);
  dim3 gQKV((NTOK / 128) * (DD / 64));
  // 3) Q/K/V projections + RoPE/reshape
  gemm_wmma_kernel<1, 0><<<gQKV, 256, 0, stream>>>(xa, wqh, bq, nullptr, qtmp, NTOK, DD, DD);
  reshape_qkv_kernel<<<rshGrid, 256, 0, stream>>>(qtmp, qbh, 0);
  gemm_wmma_kernel<1, 0><<<gQKV, 256, 0, stream>>>(xa, wkh, bk, nullptr, qtmp, NTOK, DD, DD);
  reshape_qkv_kernel<<<rshGrid, 256, 0, stream>>>(qtmp, kbh, 1);
  gemm_wmma_kernel<1, 0><<<gQKV, 256, 0, stream>>>(xa, wvh, bv, nullptr, qtmp, NTOK, DD, DD);
  reshape_qkv_kernel<<<rshGrid, 256, 0, stream>>>(qtmp, vt, 2);

  // 4) flash attention (query rows >= M only)
  attention_kernel<<<(BB * HH * (TT / 16)) / 4, 128, 0, stream>>>(qbh, kbh, vt, oh);

  // 5) Wo projection, then residual + LN
  gemm_wmma_kernel<0, 0><<<dim3((N2 / 128) * (DD / 64)), 256, 0, stream>>>(
      oh, woh, bo, nullptr, yattn, N2, DD, DD);
  residual_ln_kernel<<<N2, 256, 0, stream>>>(yattn, xfull, g_mlp, be_mlp, x2, xn);

  // 6) MLP: squared-relu then down-proj with fused residual into d_out
  gemm_wmma_kernel<1, 1><<<dim3((N2 / 128) * (DFF / 64)), 256, 0, stream>>>(
      xn, w1h, b1, nullptr, h1, N2, DFF, DD);
  gemm_wmma_kernel<0, 0><<<dim3((N2 / 128) * (DD / 64)), 256, 0, stream>>>(
      h1, w2h, b2, x2, (float*)d_out, N2, DD, DFF);
}
